// UberCRSN_24902220382597
// MI455X (gfx1250) — compile-verified, matrix operand order checked by reading, not run on hardware
//
#include <hip/hip_runtime.h>

// ---------------- problem constants ----------------
#define B_      2
#define S_      2048
#define D_      128
#define V_      32000
#define NSYM_   1024
#define KSTK_   16
#define NT_     (B_*S_)          // 4096 tokens
#define TD_     256              // 2*D
#define SCALE_  0.08838834764831845f
#define ACT_TH_ 0.99f
#define PONDER_ 1e-3f
#define COMMIT_ 0.25f
#define EPS_    1e-6f

typedef __attribute__((ext_vector_type(16))) _Float16 v16h;
typedef __attribute__((ext_vector_type(8)))  float    v8f;

static __device__ __forceinline__ v8f vzero() { v8f r = {}; return r; }

static __device__ __forceinline__ v8f wmma16(v16h a, v16h b, v8f c) {
  // D = A(16x32 f16) * B(32x16 f16) + C(16x16 f32)
  return __builtin_amdgcn_wmma_f32_16x16x32_f16(false, a, false, b, (short)0, c, false, false);
}

static __device__ __forceinline__ v16h hneg(v16h x) {
  v16h y;
#pragma unroll
  for (int i = 0; i < 16; ++i) y[i] = -x[i];
  return y;
}

// A fragment: 16(M) x 32(K), source row-major p[m*ld + k].
// Lane<16: M=lane,   K = 0..7 & 16..23 ; Lane>=16: M=lane-16, K = 8..15 & 24..31
static __device__ __forceinline__ v16h load_a16(const _Float16* p, int ld) {
  int lane = threadIdx.x & 31;
  const _Float16* r = p + (size_t)(lane & 15) * ld + ((lane >> 4) << 3);
  v16h a;
#pragma unroll
  for (int i = 0; i < 8; ++i) a[i] = r[i];
#pragma unroll
  for (int i = 0; i < 8; ++i) a[8 + i] = r[16 + i];
  return a;
}

// B fragment: 32(K) x 16(N), source p[k*ld + n]  (e.g. V matrix: key-major)
static __device__ __forceinline__ v16h load_b16(const _Float16* p, int ld) {
  int lane = threadIdx.x & 31;
  int n = lane & 15, kh = (lane >> 4) << 3;
  v16h b;
#pragma unroll
  for (int i = 0; i < 8; ++i) b[i] = p[(size_t)(kh + i) * ld + n];
#pragma unroll
  for (int i = 0; i < 8; ++i) b[8 + i] = p[(size_t)(kh + 16 + i) * ld + n];
  return b;
}

// B fragment with transposed source: B[k][n] = p[n*ld + k]  (weight W[o][k] used as out = x @ W^T)
static __device__ __forceinline__ v16h load_bt16(const _Float16* p, int ld) {
  int lane = threadIdx.x & 31;
  int n = lane & 15, kh = (lane >> 4) << 3;
  const _Float16* r = p + (size_t)n * ld + kh;
  v16h b;
#pragma unroll
  for (int i = 0; i < 8; ++i) b[i] = r[i];
#pragma unroll
  for (int i = 0; i < 8; ++i) b[8 + i] = r[16 + i];
  return b;
}

// C/D store: lane<16 rows 0..7, lane>=16 rows 8..15, col = lane&15
static __device__ __forceinline__ void store_cf(float* p, int ld, v8f c) {
  int lane = threadIdx.x & 31;
  int n = lane & 15, mb = (lane >> 4) << 3;
#pragma unroll
  for (int i = 0; i < 8; ++i) p[(size_t)(mb + i) * ld + n] = c[i];
}
static __device__ __forceinline__ void store_ch(_Float16* p, int ld, v8f c) {
  int lane = threadIdx.x & 31;
  int n = lane & 15, mb = (lane >> 4) << 3;
#pragma unroll
  for (int i = 0; i < 8; ++i) p[(size_t)(mb + i) * ld + n] = (_Float16)c[i];
}

// ---------------- utility kernels ----------------
__global__ void k_zerof(float* p, long long n) {
  long long i = (long long)blockIdx.x * blockDim.x + threadIdx.x;
  long long st = (long long)gridDim.x * blockDim.x;
  for (; i < n; i += st) p[i] = 0.f;
}

__global__ void k_cvt(const float* s, _Float16* d, long long n) {
  long long i = (long long)blockIdx.x * blockDim.x + threadIdx.x;
  long long st = (long long)gridDim.x * blockDim.x;
  for (; i < n; i += st) d[i] = (_Float16)s[i];
}

__global__ void k_fillptr(float* ptr) {
  int i = blockIdx.x * blockDim.x + threadIdx.x;
  if (i < NT_ * KSTK_) ptr[i] = ((i & (KSTK_ - 1)) == 0) ? 1.f : 0.f;
}

__global__ void k_embed(const int* idx, const float* mag, const float* ph,
                        float* zr, float* zi) {
  int i = blockIdx.x * blockDim.x + threadIdx.x;
  if (i >= NT_ * D_) return;
  int t = i >> 7, f = i & 127;
  int v = idx[t];
  float r = mag[(size_t)v * D_ + f], th = ph[(size_t)v * D_ + f];
  zr[i] = r * cosf(th);
  zi[i] = r * sinf(th);
}

// per-row (256-wide) f32->f16 convert + squared-norm; one wave per row
__global__ void k_rownorm_cvt(const float* src, _Float16* dst, float* norm, int rows) {
  int w = (blockIdx.x * blockDim.x + threadIdx.x) >> 5;
  int lane = threadIdx.x & 31;
  if (w >= rows) return;
  const float* r = src + (size_t)w * TD_;
  float acc = 0.f;
#pragma unroll
  for (int j = 0; j < 8; ++j) {
    int c = lane + 32 * j;
    float v = r[c];
    dst[(size_t)w * TD_ + c] = (_Float16)v;
    acc += v * v;
  }
#pragma unroll
  for (int off = 16; off; off >>= 1) acc += __shfl_xor(acc, off, 32);
  if (lane == 0) norm[w] = acc;
}

__global__ void k_clipzf(const float* wr, const float* wi, float* zf) {
  int i = blockIdx.x * blockDim.x + threadIdx.x;
  if (i >= NT_ * TD_) return;
  int t = i >> 8, c = i & 255;
  float v = (c < 128) ? wr[(size_t)t * D_ + c] : wi[(size_t)t * D_ + (c - 128)];
  zf[i] = fminf(5.f, fmaxf(-5.f, v));
}

// ---------------- complex linear + LN + tanh (WMMA) ----------------
__global__ __launch_bounds__(256) void k_clinear(
    const float* zr, const float* zi, const _Float16* hwr, const _Float16* hwi,
    const float* lng, const float* lnb, _Float16* htr, _Float16* hti) {
  __shared__ _Float16 szr[16][136], szi[16][136];
  __shared__ float fbuf[16][264];
  __shared__ float red[16][16];
  __shared__ float smu[16], srv[16];
  int t0 = blockIdx.x * 16;
  int tid = threadIdx.x;
  for (int e = tid; e < 16 * 128; e += 256) {
    int r = e >> 7, c = e & 127;
    szr[r][c] = (_Float16)zr[(size_t)(t0 + r) * D_ + c];
    szi[r][c] = (_Float16)zi[(size_t)(t0 + r) * D_ + c];
  }
  __syncthreads();
  int w = tid >> 5;
  v8f accr = vzero(), acci = vzero();
#pragma unroll
  for (int kb = 0; kb < 4; ++kb) {
    v16h ar = load_a16(&szr[0][0] + kb * 32, 136);
    v16h ai = load_a16(&szi[0][0] + kb * 32, 136);
    v16h br = load_bt16(hwr + (size_t)(w * 16) * D_ + kb * 32, D_);
    v16h bi = load_bt16(hwi + (size_t)(w * 16) * D_ + kb * 32, D_);
    v16h bin = hneg(bi);
    accr = wmma16(ar, br, accr);   // zr @ wr^T
    accr = wmma16(ai, bin, accr);  // - zi @ wi^T
    acci = wmma16(ai, br, acci);   // zi @ wr^T
    acci = wmma16(ar, bi, acci);   // + zr @ wi^T
  }
  store_cf(&fbuf[0][0] + w * 16, 264, accr);
  store_cf(&fbuf[0][0] + 128 + w * 16, 264, acci);
  __syncthreads();
  // fused layernorm over 256 features + tanh
  int row = tid >> 4, sub = tid & 15;
  float part = 0.f;
#pragma unroll
  for (int j = 0; j < 16; ++j) part += fbuf[row][sub + 16 * j];
  red[row][sub] = part;
  __syncthreads();
  if (sub == 0) { float m = 0.f; for (int k2 = 0; k2 < 16; ++k2) m += red[row][k2]; smu[row] = m * (1.f / 256.f); }
  __syncthreads();
  float mu = smu[row];
  part = 0.f;
#pragma unroll
  for (int j = 0; j < 16; ++j) { float d = fbuf[row][sub + 16 * j] - mu; part += d * d; }
  red[row][sub] = part;
  __syncthreads();
  if (sub == 0) { float v = 0.f; for (int k2 = 0; k2 < 16; ++k2) v += red[row][k2]; srv[row] = rsqrtf(v * (1.f / 256.f) + 1e-5f); }
  __syncthreads();
  float rv = srv[row];
  int token = t0 + row;
#pragma unroll
  for (int j = 0; j < 16; ++j) {
    int c = sub + 16 * j;
    float val = tanhf((fbuf[row][c] - mu) * rv * lng[c] + lnb[c]);
    if (c < 128) htr[(size_t)token * D_ + c] = (_Float16)val;
    else         hti[(size_t)token * D_ + (c - 128)] = (_Float16)val;
  }
}

// ---------------- q/k/v complex projections (WMMA) ----------------
__global__ __launch_bounds__(256) void k_qkv(
    const _Float16* htr, const _Float16* hti,
    const _Float16* wqr, const _Float16* wqi, const _Float16* wkr, const _Float16* wki,
    const _Float16* wvr, const _Float16* wvi,
    _Float16* hqf, _Float16* hkf, _Float16* hvr, _Float16* hvi) {
  int t0 = blockIdx.x * 16;
  int w = threadIdx.x >> 5;
  v16h atr[4], ati[4];
#pragma unroll
  for (int kb = 0; kb < 4; ++kb) {
    atr[kb] = load_a16(htr + (size_t)t0 * D_ + kb * 32, D_);
    ati[kb] = load_a16(hti + (size_t)t0 * D_ + kb * 32, D_);
  }
  v8f qr = vzero(), qi = vzero(), kr = vzero(), ki = vzero(), vr = vzero(), vi = vzero();
#pragma unroll
  for (int kb = 0; kb < 4; ++kb) {
    const size_t wo = (size_t)(w * 16) * D_ + kb * 32;
    v16h bqr = load_bt16(wqr + wo, D_), bqi = load_bt16(wqi + wo, D_);
    v16h bkr = load_bt16(wkr + wo, D_), bki = load_bt16(wki + wo, D_);
    v16h bvr = load_bt16(wvr + wo, D_), bvi = load_bt16(wvi + wo, D_);
    v16h nqi = hneg(bqi), nki = hneg(bki), nvi = hneg(bvi);
    qr = wmma16(atr[kb], bqr, qr); qr = wmma16(ati[kb], nqi, qr);
    qi = wmma16(ati[kb], bqr, qi); qi = wmma16(atr[kb], bqi, qi);
    kr = wmma16(atr[kb], bkr, kr); kr = wmma16(ati[kb], nki, kr);
    ki = wmma16(ati[kb], bkr, ki); ki = wmma16(atr[kb], bki, ki);
    vr = wmma16(atr[kb], bvr, vr); vr = wmma16(ati[kb], nvi, vr);
    vi = wmma16(ati[kb], bvr, vi); vi = wmma16(atr[kb], bvi, vi);
  }
  store_ch(hqf + (size_t)t0 * TD_ + w * 16, TD_, qr);
  store_ch(hqf + (size_t)t0 * TD_ + 128 + w * 16, TD_, qi);
  store_ch(hkf + (size_t)t0 * TD_ + w * 16, TD_, kr);
  store_ch(hkf + (size_t)t0 * TD_ + 128 + w * 16, TD_, ki);
  store_ch(hvr + (size_t)t0 * D_ + w * 16, D_, vr);
  store_ch(hvi + (size_t)t0 * D_ + w * 16, D_, vi);
}

// ---------------- flash attention (WMMA, online softmax, key-split waves) ----------------
__global__ __launch_bounds__(256) void k_attn(
    const _Float16* hqf, const _Float16* hkf, const _Float16* hvr, const _Float16* hvi,
    float* xr, float* xi) {
  __shared__ _Float16 sp[8][16][40];          // per-wave P tile (16x32)
  __shared__ float som[16][136], soi_[16][136]; // merged outputs
  __shared__ float swm[8][16], swl[8][16];
  __shared__ float smg[16], slg[16];
  int b = blockIdx.y, qt = blockIdx.x;
  int t0 = b * S_ + qt * 16;
  int tid = threadIdx.x, w = tid >> 5, lane = tid & 31;
  int half = lane >> 4, n = lane & 15;
  const _Float16* kbase = hkf + (size_t)b * S_ * TD_;
  const _Float16* vrb = hvr + (size_t)b * S_ * D_;
  const _Float16* vib = hvi + (size_t)b * S_ * D_;
  for (int e = tid; e < 16 * 136; e += 256) { (&som[0][0])[e] = 0.f; (&soi_[0][0])[e] = 0.f; }
  v16h aq[8];
#pragma unroll
  for (int kb = 0; kb < 8; ++kb) aq[kb] = load_a16(hqf + (size_t)t0 * TD_ + kb * 32, TD_);
  v8f orf[8], oif[8];
#pragma unroll
  for (int i = 0; i < 8; ++i) { orf[i] = vzero(); oif[i] = vzero(); }
  float m[8], l[8];
#pragma unroll
  for (int i = 0; i < 8; ++i) { m[i] = -1e30f; l[i] = 0.f; }
  __syncthreads();
  for (int blk = 0; blk < 8; ++blk) {                 // this wave's 8 key-blocks of 32
    int key0 = (w * 8 + blk) * 32;
    v8f s0 = vzero(), s1 = vzero();
#pragma unroll
    for (int kb = 0; kb < 8; ++kb) {
      s0 = wmma16(aq[kb], load_bt16(kbase + (size_t)key0 * TD_ + kb * 32, TD_), s0);
      s1 = wmma16(aq[kb], load_bt16(kbase + (size_t)(key0 + 16) * TD_ + kb * 32, TD_), s1);
    }
    float p0[8], p1[8], cf[8];
#pragma unroll
    for (int i = 0; i < 8; ++i) {
      float a0 = s0[i] * SCALE_, a1 = s1[i] * SCALE_;
      float bm = fmaxf(a0, a1);
#pragma unroll
      for (int off = 8; off; off >>= 1) bm = fmaxf(bm, __shfl_xor(bm, off, 32));
      float mn = fmaxf(m[i], bm);
      cf[i] = __expf(m[i] - mn);
      m[i] = mn;
      p0[i] = __expf(a0 - mn); p1[i] = __expf(a1 - mn);
      float ls = p0[i] + p1[i];
#pragma unroll
      for (int off = 8; off; off >>= 1) ls += __shfl_xor(ls, off, 32);
      l[i] = l[i] * cf[i] + ls;
    }
#pragma unroll
    for (int ft = 0; ft < 8; ++ft)
#pragma unroll
      for (int i = 0; i < 8; ++i) { orf[ft][i] *= cf[i]; oif[ft][i] *= cf[i]; }
    // P (C-layout) -> row-major LDS -> A-fragment
#pragma unroll
    for (int i = 0; i < 8; ++i) {
      sp[w][half * 8 + i][n] = (_Float16)p0[i];
      sp[w][half * 8 + i][16 + n] = (_Float16)p1[i];
    }
    __syncthreads();
    v16h ap = load_a16(&sp[w][0][0], 40);
#pragma unroll
    for (int ft = 0; ft < 8; ++ft) {
      orf[ft] = wmma16(ap, load_b16(vrb + (size_t)key0 * D_ + ft * 16, D_), orf[ft]);
      oif[ft] = wmma16(ap, load_b16(vib + (size_t)key0 * D_ + ft * 16, D_), oif[ft]);
    }
    __syncthreads();
  }
  // merge 8 wave-partials
  if (n == 0) {
#pragma unroll
    for (int i = 0; i < 8; ++i) { swm[w][half * 8 + i] = m[i]; swl[w][half * 8 + i] = l[i]; }
  }
  __syncthreads();
  if (tid < 16) {
    float mg = -1e30f;
    for (int ww = 0; ww < 8; ++ww) mg = fmaxf(mg, swm[ww][tid]);
    float lg = 0.f;
    for (int ww = 0; ww < 8; ++ww) lg += swl[ww][tid] * __expf(swm[ww][tid] - mg);
    smg[tid] = mg; slg[tid] = lg + 1e-30f;
  }
  __syncthreads();
#pragma unroll
  for (int i = 0; i < 8; ++i) {
    int row = half * 8 + i;
    float f = __expf(m[i] - smg[row]);
#pragma unroll
    for (int ft = 0; ft < 8; ++ft) {
      atomicAdd(&som[row][ft * 16 + n], orf[ft][i] * f);
      atomicAdd(&soi_[row][ft * 16 + n], oif[ft][i] * f);
    }
  }
  __syncthreads();
  int row = tid >> 4, sub = tid & 15;
  float inv = 1.f / slg[row];
  int token = t0 + row;
#pragma unroll
  for (int j = 0; j < 8; ++j) {
    int c = sub + 16 * j;
    xr[(size_t)token * D_ + c] = som[row][c] * inv;
    xi[(size_t)token * D_ + c] = soi_[row][c] * inv;
  }
}

// ---------------- halt / score heads + soft stack + ACT (one wave per token) ----------------
__global__ __launch_bounds__(256) void k_stack(
    const float* xr, const float* xi, const float* hw, const float* hb,
    const float* scw, const float* scb, float* memb, float* ptrb,
    float* halt_acc, float* n_upd, float* wr_acc, float* wi_acc,
    float* zr, float* zi, int step) {
  int token = blockIdx.x * 8 + (threadIdx.x >> 5);
  int lane = threadIdx.x & 31;
  float flat[8];
#pragma unroll
  for (int j = 0; j < 8; ++j) {
    int f = lane + 32 * j;
    flat[j] = (f < 128) ? xr[(size_t)token * D_ + f] : xi[(size_t)token * D_ + (f - 128)];
  }
  float hd = 0.f;
#pragma unroll
  for (int j = 0; j < 8; ++j) hd += flat[j] * hw[lane + 32 * j];
#pragma unroll
  for (int off = 16; off; off >>= 1) hd += __shfl_xor(hd, off, 32);
  float halt = 1.f / (1.f + __expf(-(hd + hb[0])));
  float s3[3];
#pragma unroll
  for (int k2 = 0; k2 < 3; ++k2) {
    float d = 0.f;
#pragma unroll
    for (int j = 0; j < 8; ++j) d += flat[j] * scw[k2 * TD_ + lane + 32 * j];
#pragma unroll
    for (int off = 16; off; off >>= 1) d += __shfl_xor(d, off, 32);
    s3[k2] = d + scb[k2];
  }
  float mx = fmaxf(s3[0], fmaxf(s3[1], s3[2]));
  float e0 = __expf(s3[0] - mx), e1 = __expf(s3[1] - mx), e2 = __expf(s3[2] - mx);
  float es = e0 + e1 + e2;
  float push = e0 / es, pop = e1 / es, noop = e2 / es;
  // pointer: slot s lives in lane s (mirrored in 16..31)
  int slot = lane & 15;
  float ps = ptrb[(size_t)token * KSTK_ + slot];
  float up = __shfl(ps, (slot + 15) & 15, 32);
  float dn = __shfl(ps, (slot + 1) & 15, 32);
  float np = push * up + pop * dn + noop * ps;
  float sum = np;
#pragma unroll
  for (int off = 8; off; off >>= 1) sum += __shfl_xor(sum, off, 32);
  np = np / (sum + EPS_);
  float wm = push * up;
  float racc[8];
#pragma unroll
  for (int j = 0; j < 8; ++j) racc[j] = 0.f;
  for (int s = 0; s < KSTK_; ++s) {
    float wms = __shfl(wm, s, 32);
    float nps = __shfl(np, s, 32);
    size_t base = ((size_t)token * KSTK_ + s) * TD_;
#pragma unroll
    for (int j = 0; j < 8; ++j) {
      int f = lane + 32 * j;
      float mv = memb[base + f];
      mv = wms * flat[j] + mv * (1.f - wms);
      mv = fminf(3.f, fmaxf(-3.f, mv));
      memb[base + f] = mv;
      racc[j] += mv * nps;
    }
  }
  if (lane < KSTK_) ptrb[(size_t)token * KSTK_ + slot] = np;
  float ha = halt_acc[token];
  float alive = (ha < ACT_TH_) ? 1.f : 0.f;
  int wh = ((ha + halt) >= ACT_TH_) || (step == 3);
  float wgt = alive * (wh ? (1.f - ha) : halt);
#pragma unroll
  for (int j = 0; j < 8; ++j) {
    int f = lane + 32 * j;
    float av = flat[j] + racc[j];
    if (f < 128) {
      wr_acc[(size_t)token * D_ + f] += wgt * av;
      zr[(size_t)token * D_ + f] = av;
    } else {
      int c = f - 128;
      wi_acc[(size_t)token * D_ + c] += wgt * av;
      zi[(size_t)token * D_ + c] = av;
    }
  }
  if (lane == 0) { halt_acc[token] = ha + wgt; n_upd[token] += alive; }
}

// ---------------- vector quantization (WMMA distance GEMM + argmin) ----------------
__global__ __launch_bounds__(256) void k_vq(
    const _Float16* hzf, const _Float16* hcb, const float* znorm, const float* cbnorm,
    const float* zf, const float* codebook, _Float16* hzq, float* accum) {
  __shared__ float scv[8][16];
  __shared__ int scc[8][16];
  __shared__ int swidx[16];
  __shared__ float red[16][16];
  int t0 = blockIdx.x * 16;
  int tid = threadIdx.x, w = tid >> 5, lane = tid & 31;
  int half = lane >> 4, n = lane & 15;
  v16h az[8];
#pragma unroll
  for (int kb = 0; kb < 8; ++kb) az[kb] = load_a16(hzf + (size_t)t0 * TD_ + kb * 32, TD_);
  float zn[8];
#pragma unroll
  for (int i = 0; i < 8; ++i) zn[i] = znorm[t0 + half * 8 + i];
  float bv[8]; int bc[8];
#pragma unroll
  for (int i = 0; i < 8; ++i) { bv[i] = 3.4e38f; bc[i] = 0; }
  for (int ct = 0; ct < 8; ++ct) {                     // this wave's 128 codes
    int code = w * 128 + ct * 16 + n;
    float cn = cbnorm[code];
    v8f acc = vzero();
#pragma unroll
    for (int kb = 0; kb < 8; ++kb)
      acc = wmma16(az[kb], load_bt16(hcb + (size_t)(w * 128 + ct * 16) * TD_ + kb * 32, TD_), acc);
#pragma unroll
    for (int i = 0; i < 8; ++i) {
      float d = zn[i] + cn - 2.f * acc[i];
      if (d < bv[i]) { bv[i] = d; bc[i] = code; }
    }
  }
#pragma unroll
  for (int off = 8; off; off >>= 1) {
#pragma unroll
    for (int i = 0; i < 8; ++i) {
      float ov = __shfl_xor(bv[i], off, 32);
      int   oc = __shfl_xor(bc[i], off, 32);
      if (ov < bv[i]) { bv[i] = ov; bc[i] = oc; }
    }
  }
  if (n == 0) {
#pragma unroll
    for (int i = 0; i < 8; ++i) { scv[w][half * 8 + i] = bv[i]; scc[w][half * 8 + i] = bc[i]; }
  }
  __syncthreads();
  if (tid < 16) {
    float best = 3.4e38f; int bi = 0;
    for (int ww = 0; ww < 8; ++ww) if (scv[ww][tid] < best) { best = scv[ww][tid]; bi = scc[ww][tid]; }
    swidx[tid] = bi;
  }
  __syncthreads();
  int row = tid >> 4, sub = tid & 15;
  int token = t0 + row, ci = swidx[row];
  float part = 0.f;
#pragma unroll
  for (int j = 0; j < 16; ++j) {
    int c = sub + 16 * j;
    float q = codebook[(size_t)ci * TD_ + c];
    float z = zf[(size_t)token * TD_ + c];
    float d = q - z;
    part += d * d;
    hzq[(size_t)token * TD_ + c] = (_Float16)q;  // straight-through forward value
  }
  red[row][sub] = part;
  __syncthreads();
  if (tid == 0) {
    float s = 0.f;
    for (int e = 0; e < 256; ++e) s += (&red[0][0])[e];
    atomicAdd(accum, s);
  }
}

// ---------------- decoder GEMM: (4096x256) @ (256x32000) + bias ----------------
__global__ __launch_bounds__(256) void k_decode(
    const _Float16* hzq, const _Float16* hdw, const float* decb, float* out) {
  int t0 = blockIdx.y * 16;
  int w = threadIdx.x >> 5, lane = threadIdx.x & 31;
  int n = lane & 15, mb = (lane >> 4) << 3;
  int col0 = blockIdx.x * 128 + w * 16;
  v8f acc = vzero();
#pragma unroll
  for (int kb = 0; kb < 8; ++kb) {
    v16h a = load_a16(hzq + (size_t)t0 * TD_ + kb * 32, TD_);
    v16h bb = load_bt16(hdw + (size_t)col0 * TD_ + kb * 32, TD_);
    acc = wmma16(a, bb, acc);
  }
  int col = col0 + n;
  float bias = decb[col];
#pragma unroll
  for (int i = 0; i < 8; ++i)
    out[(size_t)(t0 + mb + i) * V_ + col] = acc[i] + bias;
}

__global__ void k_aux(const float* n_upd, const float* accum, float* out) {
  __shared__ float red[256];
  float s = 0.f;
  for (int i = threadIdx.x; i < NT_; i += 256) s += n_upd[i];
  red[threadIdx.x] = s;
  __syncthreads();
  if (threadIdx.x == 0) {
    float t = 0.f;
    for (int e = 0; e < 256; ++e) t += red[e];
    float vq = accum[0] * (1.f + COMMIT_) / ((float)NT_ * (float)TD_);
    out[(size_t)NT_ * V_] = vq + PONDER_ * (t / (float)NT_);
  }
}

// ---------------- host launcher ----------------
extern "C" void kernel_launch(void* const* d_in, const int* in_sizes, int n_in,
                              void* d_out, int out_size, void* d_ws, size_t ws_size,
                              hipStream_t stream) {
  (void)in_sizes; (void)n_in; (void)out_size; (void)ws_size;
  const int*   idx     = (const int*)d_in[0];
  const float* emb_mag = (const float*)d_in[1];
  const float* emb_ph  = (const float*)d_in[2];
  const float* lin_wr  = (const float*)d_in[3];
  const float* lin_wi  = (const float*)d_in[4];
  const float* ln_g    = (const float*)d_in[5];
  const float* ln_b    = (const float*)d_in[6];
  const float* halt_w  = (const float*)d_in[7];
  const float* halt_b  = (const float*)d_in[8];
  const float* sc_w    = (const float*)d_in[9];
  const float* sc_b    = (const float*)d_in[10];
  const float* aqr     = (const float*)d_in[11];
  const float* aqi     = (const float*)d_in[12];
  const float* akr     = (const float*)d_in[13];
  const float* aki     = (const float*)d_in[14];
  const float* avr     = (const float*)d_in[15];
  const float* avi     = (const float*)d_in[16];
  const float* codebook= (const float*)d_in[17];
  const float* dec_w   = (const float*)d_in[18];
  const float* dec_b   = (const float*)d_in[19];
  float* out = (float*)d_out;

  char* cur = (char*)d_ws;
  auto af = [&](size_t nel) { float* p = (float*)cur; cur += ((nel * 4 + 255) & ~(size_t)255); return p; };
  auto ah = [&](size_t nel) { _Float16* p = (_Float16*)cur; cur += ((nel * 2 + 255) & ~(size_t)255); return p; };

  float* zr = af((size_t)NT_ * D_);
  float* zi = af((size_t)NT_ * D_);
  float* xr = af((size_t)NT_ * D_);
  float* xi = af((size_t)NT_ * D_);
  float* zf = af((size_t)NT_ * TD_);
  float* znorm = af(NT_);
  float* cbnorm = af(NSYM_);
  float* ptrb = af((size_t)NT_ * KSTK_);
  size_t state_n = (size_t)NT_ * KSTK_ * TD_ + NT_ + NT_ + (size_t)NT_ * D_ + (size_t)NT_ * D_ + 64;
  float* state = af(state_n);
  float* memb = state;
  float* halt_acc = memb + (size_t)NT_ * KSTK_ * TD_;
  float* n_upd = halt_acc + NT_;
  float* wr_acc = n_upd + NT_;
  float* wi_acc = wr_acc + (size_t)NT_ * D_;
  float* accum = wi_acc + (size_t)NT_ * D_;

  _Float16* h_tr = ah((size_t)NT_ * D_);
  _Float16* h_ti = ah((size_t)NT_ * D_);
  _Float16* h_qf = ah((size_t)NT_ * TD_);
  _Float16* h_kf = ah((size_t)NT_ * TD_);
  _Float16* h_vr = ah((size_t)NT_ * D_);
  _Float16* h_vi = ah((size_t)NT_ * D_);
  _Float16* h_zf = ah((size_t)NT_ * TD_);
  _Float16* h_zq = ah((size_t)NT_ * TD_);
  _Float16* h_lwr = ah((size_t)D_ * D_);
  _Float16* h_lwi = ah((size_t)D_ * D_);
  _Float16* h_wqr = ah((size_t)D_ * D_);
  _Float16* h_wqi = ah((size_t)D_ * D_);
  _Float16* h_wkr = ah((size_t)D_ * D_);
  _Float16* h_wki = ah((size_t)D_ * D_);
  _Float16* h_wvr = ah((size_t)D_ * D_);
  _Float16* h_wvi = ah((size_t)D_ * D_);
  _Float16* h_cb  = ah((size_t)NSYM_ * TD_);
  _Float16* h_dw  = ah((size_t)V_ * TD_);

  // init state
  k_zerof<<<4096, 256, 0, stream>>>(state, (long long)state_n);
  k_fillptr<<<(NT_ * KSTK_ + 255) / 256, 256, 0, stream>>>(ptrb);
  k_embed<<<(NT_ * D_ + 255) / 256, 256, 0, stream>>>(idx, emb_mag, emb_ph, zr, zi);
  // weight conversions to f16 (done every call; deterministic)
  long long wn = (long long)D_ * D_;
  k_cvt<<<64, 256, 0, stream>>>(lin_wr, h_lwr, wn);
  k_cvt<<<64, 256, 0, stream>>>(lin_wi, h_lwi, wn);
  k_cvt<<<64, 256, 0, stream>>>(aqr, h_wqr, wn);
  k_cvt<<<64, 256, 0, stream>>>(aqi, h_wqi, wn);
  k_cvt<<<64, 256, 0, stream>>>(akr, h_wkr, wn);
  k_cvt<<<64, 256, 0, stream>>>(aki, h_wki, wn);
  k_cvt<<<64, 256, 0, stream>>>(avr, h_wvr, wn);
  k_cvt<<<64, 256, 0, stream>>>(avi, h_wvi, wn);
  k_cvt<<<4096, 256, 0, stream>>>(dec_w, h_dw, (long long)V_ * TD_);
  k_rownorm_cvt<<<NSYM_ / 8, 256, 0, stream>>>(codebook, h_cb, cbnorm, NSYM_);

  for (int step = 0; step < 4; ++step) {
    k_clinear<<<NT_ / 16, 256, 0, stream>>>(zr, zi, h_lwr, h_lwi, ln_g, ln_b, h_tr, h_ti);
    k_qkv<<<NT_ / 16, 256, 0, stream>>>(h_tr, h_ti, h_wqr, h_wqi, h_wkr, h_wki, h_wvr, h_wvi,
                                        h_qf, h_kf, h_vr, h_vi);
    dim3 ga(S_ / 16, B_);
    k_attn<<<ga, 256, 0, stream>>>(h_qf, h_kf, h_vr, h_vi, xr, xi);
    k_stack<<<NT_ / 8, 256, 0, stream>>>(xr, xi, halt_w, halt_b, sc_w, sc_b, memb, ptrb,
                                         halt_acc, n_upd, wr_acc, wi_acc, zr, zi, step);
  }
  k_clipzf<<<(NT_ * TD_ + 255) / 256, 256, 0, stream>>>(wr_acc, wi_acc, zf);
  k_rownorm_cvt<<<NT_ / 8, 256, 0, stream>>>(zf, h_zf, znorm, NT_);
  k_vq<<<NT_ / 16, 256, 0, stream>>>(h_zf, h_cb, znorm, cbnorm, zf, codebook, h_zq, accum);
  dim3 gd(V_ / 128, NT_ / 16);
  k_decode<<<gd, 256, 0, stream>>>(h_zq, h_dw, dec_b, out);
  k_aux<<<1, 256, 0, stream>>>(n_upd, accum, out);
}
